// Attention_9612136808713
// MI455X (gfx1250) — matrix operand order repaired and checked by reading, not compiled, not run on hardware
//
#include <hip/hip_runtime.h>
#include <hip/hip_bf16.h>

#define HEADS 16
#define BATCH 2
#define SEQ   2048
#define DIMF  1024
#define DHEAD 64
#define ATT_SCALE 0.125f

typedef __bf16 bf16_t;
typedef __attribute__((ext_vector_type(16))) __bf16 v16bf;
typedef __attribute__((ext_vector_type(8)))  float  v8f;
typedef __attribute__((ext_vector_type(4)))  unsigned int v4u;

union FragU { v16bf v; v4u q[2]; };

// A operand (16x32 bf16): lane holds row M=lane&15.
// lane<16: K = 0..7 and 16..23 ; lane>=16: K = 8..15 and 24..31.
__device__ inline v16bf load_frag_a(const bf16_t* p, int lane) {
  FragU f;
  const int o = (lane & 16) >> 1;            // 0 or 8
  f.q[0] = *(const v4u*)(p + o);
  f.q[1] = *(const v4u*)(p + o + 16);
  return f.v;
}
// B operand (32x16 bf16): lane holds column N=lane&15.
// lane<16: K = 0..7 and 16..23 ; lane>=16: K = 8..15 and 24..31 (K-contiguous source rows).
__device__ inline v16bf load_frag_b(const bf16_t* p, int lane) {
  FragU f;
  const int o = (lane & 16) >> 1;            // 0 or 8
  f.q[0] = *(const v4u*)(p + o);
  f.q[1] = *(const v4u*)(p + o + 16);
  return f.v;
}

__device__ inline v8f wmma_bf16(v16bf a, v16bf b, v8f c) {
  return __builtin_amdgcn_wmma_f32_16x16x32_bf16(false, a, false, b, (short)0, c,
                                                 false, false);
}

// ---------------- cast / transpose / zero helpers ----------------
__global__ void cast_f32_bf16(const float* __restrict__ s, bf16_t* __restrict__ d, int n) {
  for (int i = blockIdx.x * blockDim.x + threadIdx.x; i < n; i += gridDim.x * blockDim.x)
    d[i] = (bf16_t)s[i];
}
__global__ void castT_f32_bf16(const float* __restrict__ s, bf16_t* __restrict__ d,
                               int rows, int cols) {
  int n = rows * cols;
  for (int i = blockIdx.x * blockDim.x + threadIdx.x; i < n; i += gridDim.x * blockDim.x) {
    int r = i / cols, c = i % cols;
    d[(size_t)c * rows + r] = (bf16_t)s[i];
  }
}
__global__ void zero_bf16(bf16_t* __restrict__ d, int n) {
  for (int i = blockIdx.x * blockDim.x + threadIdx.x; i < n; i += gridDim.x * blockDim.x)
    d[i] = (bf16_t)0.0f;
}

// ---------------- generic bf16 WMMA GEMM:  C = A @ WT^T + bias -----------
// Wave tile 32x64, register double-buffered over K.
// A: [M,K] bf16 row-major.  WT: [N,K] bf16 (pre-transposed weight).
// mode 0: bf16 out, ld=1024.     mode 1: kv split (k row-major + v transposed)
// mode 2: bf16 out, ld=64 (p).   mode 3: fp32 out, ld=1024 (final).
__global__ __launch_bounds__(128) void gemm_bf16(
    const bf16_t* __restrict__ A, const bf16_t* __restrict__ WT,
    const float* __restrict__ bias, int K, int mode,
    bf16_t* __restrict__ out_b, float* __restrict__ out_f, bf16_t* __restrict__ out_v) {
  const int lane = threadIdx.x & 31;
  const int w = threadIdx.x >> 5;
  const int row0 = blockIdx.y * 128 + w * 32;
  const int cb = blockIdx.x * 64;
  const int ln15 = lane & 15;
  const int hb = (lane & 16) >> 1;

  const v8f zero = {0.f, 0.f, 0.f, 0.f, 0.f, 0.f, 0.f, 0.f};
  v8f acc[2][4];
#pragma unroll
  for (int rr = 0; rr < 2; ++rr)
#pragma unroll
    for (int tt = 0; tt < 4; ++tt) acc[rr][tt] = zero;

  const bf16_t* ar0 = A + (size_t)(row0 + ln15) * K;
  const bf16_t* ar1 = ar0 + (size_t)16 * K;
  const bf16_t* wb = WT + (size_t)(cb + ln15) * K;

  // prologue loads
  v16bf a0 = load_frag_a(ar0, lane);
  v16bf a1 = load_frag_a(ar1, lane);
  v16bf b0 = load_frag_b(wb, lane);
  v16bf b1 = load_frag_b(wb + (size_t)16 * K, lane);
  v16bf b2 = load_frag_b(wb + (size_t)32 * K, lane);
  v16bf b3 = load_frag_b(wb + (size_t)48 * K, lane);

  for (int k0 = 0; k0 < K - 32; k0 += 32) {
    const int kn = k0 + 32;
    // double-buffer: issue next k-step's loads before consuming current frags
    v16bf a0n = load_frag_a(ar0 + kn, lane);
    v16bf a1n = load_frag_a(ar1 + kn, lane);
    v16bf b0n = load_frag_b(wb + kn, lane);
    v16bf b1n = load_frag_b(wb + (size_t)16 * K + kn, lane);
    v16bf b2n = load_frag_b(wb + (size_t)32 * K + kn, lane);
    v16bf b3n = load_frag_b(wb + (size_t)48 * K + kn, lane);

    acc[0][0] = wmma_bf16(a0, b0, acc[0][0]);
    acc[1][0] = wmma_bf16(a1, b0, acc[1][0]);
    acc[0][1] = wmma_bf16(a0, b1, acc[0][1]);
    acc[1][1] = wmma_bf16(a1, b1, acc[1][1]);
    acc[0][2] = wmma_bf16(a0, b2, acc[0][2]);
    acc[1][2] = wmma_bf16(a1, b2, acc[1][2]);
    acc[0][3] = wmma_bf16(a0, b3, acc[0][3]);
    acc[1][3] = wmma_bf16(a1, b3, acc[1][3]);

    a0 = a0n; a1 = a1n; b0 = b0n; b1 = b1n; b2 = b2n; b3 = b3n;
  }
  // tail k-step
  acc[0][0] = wmma_bf16(a0, b0, acc[0][0]);
  acc[1][0] = wmma_bf16(a1, b0, acc[1][0]);
  acc[0][1] = wmma_bf16(a0, b1, acc[0][1]);
  acc[1][1] = wmma_bf16(a1, b1, acc[1][1]);
  acc[0][2] = wmma_bf16(a0, b2, acc[0][2]);
  acc[1][2] = wmma_bf16(a1, b2, acc[1][2]);
  acc[0][3] = wmma_bf16(a0, b3, acc[0][3]);
  acc[1][3] = wmma_bf16(a1, b3, acc[1][3]);

#pragma unroll
  for (int tt = 0; tt < 4; ++tt) {
    const int col = cb + tt * 16 + ln15;
    const float bv = bias[col];
#pragma unroll
    for (int rr = 0; rr < 2; ++rr) {
#pragma unroll
      for (int r = 0; r < 8; ++r) {
        const int row = row0 + rr * 16 + r + hb;
        const float val = acc[rr][tt][r] + bv;
        if (mode == 0) {
          out_b[(size_t)row * DIMF + col] = (bf16_t)val;
        } else if (mode == 1) {
          if (col < DIMF) {
            out_b[(size_t)row * DIMF + col] = (bf16_t)val;   // k
          } else {
            const int c2 = col - DIMF, h = c2 >> 6, dd = c2 & 63;
            const int bi = row >> 11, j = row & (SEQ - 1);
            out_v[((size_t)(bi * HEADS + h) * DHEAD + dd) * SEQ + j] = (bf16_t)val; // vT
          }
        } else if (mode == 2) {
          out_b[(size_t)row * DHEAD + col] = (bf16_t)val;
        } else {
          out_f[(size_t)row * DIMF + col] = val;
        }
      }
    }
  }
}

// ---------------- fused rel-pos flash attention ----------------
// Q,Km : [B*SEQ, DIM] bf16 (head h occupies cols h*64..h*64+63)
// Vt   : [B,H,DHEAD,SEQ] bf16.  Pb : [SEQ+64, DHEAD] bf16 (rows >= SEQ are zero)
// O    : [B*SEQ, DIM] bf16 (pre-Wo attention output)
__global__ __launch_bounds__(128) void attn_kernel(
    const bf16_t* __restrict__ Q, const bf16_t* __restrict__ Km,
    const bf16_t* __restrict__ Vt, const bf16_t* __restrict__ Pb,
    bf16_t* __restrict__ O) {
  __shared__ float  Gs[4][16][80];
  __shared__ bf16_t Ps[4][16][64];

  const int lane = threadIdx.x & 31;
  const int w = threadIdx.x >> 5;
  const int bh = blockIdx.y;
  const int bi = bh >> 4, h = bh & 15;
  const int i0blk = blockIdx.x * 64;
  const int i0w = i0blk + w * 16;
  const int hb = (lane & 16) >> 1;
  const int ln15 = lane & 15;

  const bf16_t* qrow = Q + (size_t)(bi * SEQ + i0w + ln15) * DIMF + h * DHEAD;
  const v16bf qa0 = load_frag_a(qrow, lane);
  const v16bf qa1 = load_frag_a(qrow + 32, lane);

  const v8f zero = {0.f, 0.f, 0.f, 0.f, 0.f, 0.f, 0.f, 0.f};
  v8f oac[4];
#pragma unroll
  for (int tb = 0; tb < 4; ++tb) oac[tb] = zero;
  float m[8], l[8];
#pragma unroll
  for (int r = 0; r < 8; ++r) { m[r] = -3.0e38f; l[r] = 0.0f; }

  const bf16_t* kbase = Km + (size_t)bi * SEQ * DIMF + h * DHEAD;
  const bf16_t* vbase = Vt + (size_t)(bi * HEADS + h) * DHEAD * SEQ;

  for (int j0 = 0; j0 <= i0blk; j0 += 64) {
    const int u0w = SEQ - 16 - i0w + j0;

    // ---- stage all positional + key fragments up front (long load->use) ----
    v16bf pf[5][2];
#pragma unroll
    for (int t = 0; t < 5; ++t) {
      const bf16_t* prow = Pb + (size_t)(u0w + t * 16 + ln15) * DHEAD;
      pf[t][0] = load_frag_b(prow, lane);
      pf[t][1] = load_frag_b(prow + 32, lane);
    }
    v16bf kf[4][2];
#pragma unroll
    for (int tt = 0; tt < 4; ++tt) {
      const bf16_t* krow = kbase + (size_t)(j0 + tt * 16 + ln15) * DIMF;
      kf[tt][0] = load_frag_b(krow, lane);
      kf[tt][1] = load_frag_b(krow + 32, lane);
    }
    // prefetch next j-block's K/V rows into cache (global_prefetch_b8)
    if (j0 + 64 <= i0blk) {
      __builtin_prefetch(kbase + (size_t)(j0 + 64 + ln15) * DIMF, 0, 0);
      __builtin_prefetch(vbase + (size_t)ln15 * SEQ + j0 + 64, 0, 0);
    }

    // G[i,u] = q_i . p[u0w+u]   (16 x 80; rel-shifted bias = G[i, jj+15-mi])
#pragma unroll
    for (int t = 0; t < 5; ++t) {
      v8f g = zero;
      g = wmma_bf16(qa0, pf[t][0], g);
      g = wmma_bf16(qa1, pf[t][1], g);
#pragma unroll
      for (int r = 0; r < 8; ++r) Gs[w][r + hb][t * 16 + ln15] = g[r];
    }
    asm volatile("s_wait_dscnt 0" ::: "memory");

    // S = Q . K^T
    v8f sc[4];
#pragma unroll
    for (int tt = 0; tt < 4; ++tt) {
      v8f s = zero;
      s = wmma_bf16(qa0, kf[tt][0], s);
      s = wmma_bf16(qa1, kf[tt][1], s);
      sc[tt] = s;
    }

    // issue V fragment loads now; they overlap the softmax VALU chain
    v16bf vf[4][2];
#pragma unroll
    for (int tb = 0; tb < 4; ++tb) {
      const bf16_t* vrow = vbase + (size_t)(tb * 16 + ln15) * SEQ + j0;
      vf[tb][0] = load_frag_b(vrow, lane);
      vf[tb][1] = load_frag_b(vrow + 32, lane);
    }

    // add shifted bias, scale, causal mask
#pragma unroll
    for (int tt = 0; tt < 4; ++tt) {
#pragma unroll
      for (int r = 0; r < 8; ++r) {
        const int mi = r + hb;
        const int jj = tt * 16 + ln15;
        float s = (sc[tt][r] + Gs[w][mi][jj + 15 - mi]) * ATT_SCALE;
        if (j0 + jj > i0w + mi) s = -1.0e30f;
        sc[tt][r] = s;
      }
    }

    // online softmax update
#pragma unroll
    for (int r = 0; r < 8; ++r) {
      float mx = fmaxf(fmaxf(sc[0][r], sc[1][r]), fmaxf(sc[2][r], sc[3][r]));
      mx = fmaxf(mx, __shfl_xor(mx, 1));
      mx = fmaxf(mx, __shfl_xor(mx, 2));
      mx = fmaxf(mx, __shfl_xor(mx, 4));
      mx = fmaxf(mx, __shfl_xor(mx, 8));
      const float mn = fmaxf(m[r], mx);
      const float alpha = __expf(m[r] - mn);
      m[r] = mn;
      l[r] *= alpha;
#pragma unroll
      for (int tb = 0; tb < 4; ++tb) oac[tb][r] *= alpha;
#pragma unroll
      for (int tt = 0; tt < 4; ++tt) {
        const float p = __expf(sc[tt][r] - mn);
        l[r] += p;
        Ps[w][r + hb][tt * 16 + ln15] = (bf16_t)p;
      }
    }
    asm volatile("s_wait_dscnt 0" ::: "memory");

    // reload probabilities as A fragments (C-layout -> A-layout via LDS)
    const bf16_t* pl = &Ps[w][ln15][0];
    FragU f0, f1;
    const int o = (lane & 16) >> 1;
    f0.q[0] = *(const v4u*)(pl + o);
    f0.q[1] = *(const v4u*)(pl + o + 16);
    f1.q[0] = *(const v4u*)(pl + 32 + o);
    f1.q[1] = *(const v4u*)(pl + 32 + o + 16);
    const v16bf pa0 = f0.v, pa1 = f1.v;

    // O += P . V
#pragma unroll
    for (int tb = 0; tb < 4; ++tb) {
      oac[tb] = wmma_bf16(pa0, vf[tb][0], oac[tb]);
      oac[tb] = wmma_bf16(pa1, vf[tb][1], oac[tb]);
    }
  }

  // normalize and store
#pragma unroll
  for (int r = 0; r < 8; ++r) {
    float s = l[r];
    s += __shfl_xor(s, 1);
    s += __shfl_xor(s, 2);
    s += __shfl_xor(s, 4);
    s += __shfl_xor(s, 8);
    l[r] = 1.0f / s;
  }
#pragma unroll
  for (int tb = 0; tb < 4; ++tb) {
#pragma unroll
    for (int r = 0; r < 8; ++r) {
      const float val = oac[tb][r] * l[r];
      O[(size_t)(bi * SEQ + i0w + r + hb) * DIMF + h * DHEAD + tb * 16 + ln15] =
          (bf16_t)val;
    }
  }
}

// ---------------- host-side orchestration ----------------
extern "C" void kernel_launch(void* const* d_in, const int* in_sizes, int n_in,
                              void* d_out, int out_size, void* d_ws, size_t ws_size,
                              hipStream_t stream) {
  (void)in_sizes; (void)n_in; (void)out_size; (void)ws_size;
  const float* x   = (const float*)d_in[0];
  const float* pe  = (const float*)d_in[1];
  const float* Wq  = (const float*)d_in[2];
  const float* bq  = (const float*)d_in[3];
  const float* Wkv = (const float*)d_in[4];
  const float* bkv = (const float*)d_in[5];
  const float* Wp  = (const float*)d_in[6];
  const float* bp  = (const float*)d_in[7];
  const float* Wo  = (const float*)d_in[8];
  const float* bo  = (const float*)d_in[9];
  float* out = (float*)d_out;

  char* ws = (char*)d_ws;
  size_t off = 0;
  auto alloc = [&](size_t elems) {
    bf16_t* p = (bf16_t*)(ws + off);
    off += (elems * sizeof(bf16_t) + 255) & ~(size_t)255;
    return p;
  };
  bf16_t* xb   = alloc((size_t)BATCH * SEQ * DIMF);
  bf16_t* peb  = alloc((size_t)SEQ * DIMF);
  bf16_t* wqT  = alloc((size_t)DIMF * DIMF);
  bf16_t* wkvT = alloc((size_t)DIMF * 2 * DIMF);
  bf16_t* wpT  = alloc((size_t)DIMF * DHEAD);
  bf16_t* woT  = alloc((size_t)DIMF * DIMF);
  bf16_t* pb   = alloc((size_t)(SEQ + 64) * DHEAD);
  bf16_t* qb   = alloc((size_t)BATCH * SEQ * DIMF);
  bf16_t* kb   = alloc((size_t)BATCH * SEQ * DIMF);
  bf16_t* vT   = alloc((size_t)BATCH * SEQ * DIMF);
  bf16_t* ob   = alloc((size_t)BATCH * SEQ * DIMF);

  const int thr = 256;
  cast_f32_bf16<<<512, thr, 0, stream>>>(x, xb, BATCH * SEQ * DIMF);
  cast_f32_bf16<<<512, thr, 0, stream>>>(pe, peb, SEQ * DIMF);
  castT_f32_bf16<<<512, thr, 0, stream>>>(Wq, wqT, DIMF, DIMF);
  castT_f32_bf16<<<512, thr, 0, stream>>>(Wkv, wkvT, DIMF, 2 * DIMF);
  castT_f32_bf16<<<64, thr, 0, stream>>>(Wp, wpT, DIMF, DHEAD);
  castT_f32_bf16<<<512, thr, 0, stream>>>(Wo, woT, DIMF, DIMF);
  zero_bf16<<<16, thr, 0, stream>>>(pb + (size_t)SEQ * DHEAD, 64 * DHEAD);

  // p = pos_emb @ Wp + bp   -> pb rows [0, SEQ)
  gemm_bf16<<<dim3(1, SEQ / 128), 128, 0, stream>>>(peb, wpT, bp, DIMF, 2, pb, nullptr, nullptr);
  // q = x @ Wq + bq
  gemm_bf16<<<dim3(DIMF / 64, BATCH * SEQ / 128), 128, 0, stream>>>(xb, wqT, bq, DIMF, 0, qb, nullptr, nullptr);
  // kv = x @ Wkv + bkv  (k row-major, v transposed)
  gemm_bf16<<<dim3(2 * DIMF / 64, BATCH * SEQ / 128), 128, 0, stream>>>(xb, wkvT, bkv, DIMF, 1, kb, nullptr, vT);
  // fused rel-pos causal flash attention
  attn_kernel<<<dim3(SEQ / 64, BATCH * HEADS), 128, 0, stream>>>(qb, kb, vT, pb, ob);
  // out = o @ Wo + bo  (fp32)
  gemm_bf16<<<dim3(DIMF / 64, BATCH * SEQ / 128), 128, 0, stream>>>(ob, woT, bo, DIMF, 3, nullptr, out, nullptr);
}
